// GATConvGRUManual_1949915152794
// MI455X (gfx1250) — compile-verified, matrix operand order checked by reading, not run on hardware
//
#include <hip/hip_runtime.h>
#include <hip/hip_bf16.h>

#define NND   20000
#define NED   320000
#define DIN   512
#define HIDC  256
#define NH    8
#define HD    32
#define ATT_SCALE 0.17677669529663687f   // 1/sqrt(32)

typedef __attribute__((ext_vector_type(16))) __bf16 v16bf;
typedef __attribute__((ext_vector_type(8)))  float  v8f;

union FragU { unsigned int u[8]; uint4 q[2]; v16bf v; };

static __device__ __forceinline__ unsigned short f2bf(float f) {
    unsigned int u = __float_as_uint(f);
    u += 0x7FFFu + ((u >> 16) & 1u);      // round-to-nearest-even
    return (unsigned short)(u >> 16);
}

static __device__ __forceinline__ void atomicMaxF(float* addr, float v) {
    // valid for buffers initialized to -inf (bit pattern 0xFF800000)
    if (v >= 0.0f) atomicMax((int*)addr, __float_as_int(v));
    else           atomicMin((unsigned int*)addr, (unsigned int)__float_as_int(v));
}

// LDS byte offset of a __shared__ object: generic pointers to LDS carry the
// allocation-relative offset in their low 32 bits (flat->LDS uses addr[31:0]).
static __device__ __forceinline__ unsigned lds_off(const void* p) {
    return (unsigned)(unsigned long long)p;
}

// ---------------------------------------------------------------- elementwise
__global__ __launch_bounds__(256) void k_concat(const float* __restrict__ x,
                                                const float* __restrict__ h,
                                                unsigned short* __restrict__ hb) {
    int i = blockIdx.x * 256 + threadIdx.x;           // over N*512
    int n = i >> 9, c = i & 511;
    float v = (c < 256) ? x[(size_t)n * 256 + c] : h[(size_t)n * 256 + (c - 256)];
    hb[i] = f2bf(v);
}

__global__ __launch_bounds__(256) void k_hcand(const float* __restrict__ x,
                                               const float* __restrict__ h,
                                               const float* __restrict__ r,
                                               unsigned short* __restrict__ hb) {
    int i = blockIdx.x * 256 + threadIdx.x;           // over N*512
    int n = i >> 9, c = i & 511;
    float v;
    if (c < 256) v = x[(size_t)n * 256 + c];
    else {
        size_t j = (size_t)n * 256 + (c - 256);
        v = r[j] * h[j];
    }
    hb[i] = f2bf(v);
}

__global__ __launch_bounds__(256) void k_w2bf(const float* __restrict__ W,
                                              unsigned short* __restrict__ Wb) {
    int i = blockIdx.x * 256 + threadIdx.x;           // over 512*256
    int k = i >> 8, n = i & 255;
    Wb[(size_t)n * DIN + k] = f2bf(W[i]);             // store transposed [n][k]
}

__global__ __launch_bounds__(256) void k_init_ms(float* __restrict__ Mx,
                                                 float* __restrict__ Sm) {
    int i = blockIdx.x * 256 + threadIdx.x;           // over N*8
    Mx[i] = __int_as_float(0xFF800000);               // -inf
    Sm[i] = 0.0f;
}

__global__ __launch_bounds__(256) void k_zero(float* __restrict__ p) {
    int i = blockIdx.x * 256 + threadIdx.x;
    p[i] = 0.0f;
}

// ---------------------------------------------------------------- WMMA GEMM
// C[N,256] = A[N,512](bf16) @ W[512,256] + bias, W pre-transposed bf16 [256][512]
// Double-buffered async (memory->LDS) pipeline: next tile's DMA overlaps WMMAs.
// LDS row stride = 40 ushorts (80 B): 16-B aligned for ds_load_b128 fragment
// reads, and conflict-free (start banks 20*m mod 64 tile all 64 banks).
__global__ __launch_bounds__(256) void k_gemm(const unsigned short* __restrict__ A,
                                              const unsigned short* __restrict__ Wt,
                                              const float* __restrict__ bias,
                                              float* __restrict__ C, int nrows) {
    __shared__ unsigned short As[2][64][40];    // 64 rows x 32 k (pad 8)
    __shared__ unsigned short Bs[2][256][40];   // 256 cols x 32 k (pad 8)

    const int tid  = threadIdx.x;
    const int wave = tid >> 5, lane = tid & 31;
    const int half = lane >> 4, l15 = lane & 15;
    const int m0 = blockIdx.x * 64;
    const int mw = (wave >> 2) * 32;         // 0 or 32
    const int nw = (wave & 3) * 64;          // 0,64,128,192

    v8f acc[2][4];
#pragma unroll
    for (int mt = 0; mt < 2; ++mt)
#pragma unroll
        for (int nt = 0; nt < 4; ++nt)
#pragma unroll
            for (int r = 0; r < 8; ++r) acc[mt][nt][r] = 0.0f;

    const int arow = tid >> 2;               // 0..63
    const int acol = (tid & 3) * 8;          // 0,8,16,24 (bf16 elems)
    const int grow = m0 + arow;
    // clamp OOB rows to a valid address: garbage rows only feed D rows that
    // are never stored (row m of A contributes only to row m of D)
    const int growc = (grow < nrows) ? grow : (nrows - 1);

    const unsigned short* gA = A  + (size_t)growc * DIN + acol;
    const unsigned short* gW = Wt + (size_t)tid   * DIN;
    const unsigned aL[2] = { lds_off(&As[0][arow][acol]), lds_off(&As[1][arow][acol]) };
    const unsigned wL[2] = { lds_off(&Bs[0][tid][0]),     lds_off(&Bs[1][tid][0]) };

    // 5 async loads per thread per tile: 1x16B for A, 4x16B for W.
    // NOTE: inst offset is added to BOTH the LDS and global addresses.
    auto issue = [&](int k0, unsigned al, unsigned wl) {
        const unsigned short* ga = gA + k0;
        const unsigned short* gw = gW + k0;
        asm volatile("global_load_async_to_lds_b128 %0, %1, off"
                     :: "v"(al), "v"(ga) : "memory");
        asm volatile("global_load_async_to_lds_b128 %0, %1, off"
                     :: "v"(wl), "v"(gw) : "memory");
        asm volatile("global_load_async_to_lds_b128 %0, %1, off offset:16"
                     :: "v"(wl), "v"(gw) : "memory");
        asm volatile("global_load_async_to_lds_b128 %0, %1, off offset:32"
                     :: "v"(wl), "v"(gw) : "memory");
        asm volatile("global_load_async_to_lds_b128 %0, %1, off offset:48"
                     :: "v"(wl), "v"(gw) : "memory");
    };

    issue(0, aL[0], wL[0]);

#pragma unroll
    for (int i = 0; i < DIN / 32; ++i) {
        const int cur = i & 1;
        if (i < DIN / 32 - 1) {
            issue((i + 1) * 32, aL[cur ^ 1], wL[cur ^ 1]);
            // async ops complete in order: <=5 outstanding => current tile landed
            asm volatile("s_wait_asynccnt 0x5" ::: "memory");
        } else {
            asm volatile("s_wait_asynccnt 0x0" ::: "memory");
        }
        __syncthreads();

        // ---- fragments: each 8-dword frag = two contiguous 16-B LDS reads ----
        // u[0..3] <-> kk = half*8 + {0,2,4,6}; u[4..7] <-> kk = 16 + half*8 + {0,2,4,6}
        FragU afr[2], bfr[4];
        const int kb = half * 8;
#pragma unroll
        for (int mt = 0; mt < 2; ++mt) {
            afr[mt].q[0] = *(const uint4*)&As[cur][mw + mt * 16 + l15][kb];
            afr[mt].q[1] = *(const uint4*)&As[cur][mw + mt * 16 + l15][16 + kb];
        }
#pragma unroll
        for (int nt = 0; nt < 4; ++nt) {
            bfr[nt].q[0] = *(const uint4*)&Bs[cur][nw + nt * 16 + l15][kb];
            bfr[nt].q[1] = *(const uint4*)&Bs[cur][nw + nt * 16 + l15][16 + kb];
        }
#pragma unroll
        for (int mt = 0; mt < 2; ++mt)
#pragma unroll
            for (int nt = 0; nt < 4; ++nt)
                acc[mt][nt] = __builtin_amdgcn_wmma_f32_16x16x32_bf16(
                    false, afr[mt].v, false, bfr[nt].v,
                    (short)0, acc[mt][nt], false, false);

        __syncthreads();   // all reads of buf[cur] done before it is refilled
    }

    // ---- store D per ISA f32 16x16 C/D layout ----
#pragma unroll
    for (int nt = 0; nt < 4; ++nt) {
        const int col = nw + nt * 16 + l15;
        const float bv = bias[col];
#pragma unroll
        for (int mt = 0; mt < 2; ++mt) {
#pragma unroll
            for (int r = 0; r < 8; ++r) {
                const int row = m0 + mw + mt * 16 + (half ? r + 8 : r);
                if (row < nrows) C[(size_t)row * 256 + col] = acc[mt][nt][r] + bv;
            }
        }
    }
}

// ---------------------------------------------------------------- edge passes
__global__ __launch_bounds__(256) void k_logits(const float* __restrict__ Q,
                                                const float* __restrict__ K,
                                                const int* __restrict__ src,
                                                const int* __restrict__ dst,
                                                float* __restrict__ elog,
                                                float* __restrict__ Mx) {
    const int e = blockIdx.x * 8 + (threadIdx.x >> 5);
    const int lane = threadIdx.x & 31;
    const int s = src[e], d = dst[e];
    const float* q = Q + (size_t)d * 256;
    const float* k = K + (size_t)s * 256;
    float p[8];
#pragma unroll
    for (int j = 0; j < 8; ++j) p[j] = q[j * 32 + lane] * k[j * 32 + lane];
#pragma unroll
    for (int m = 16; m >= 1; m >>= 1)
#pragma unroll
        for (int j = 0; j < 8; ++j) p[j] += __shfl_xor(p[j], m, 32);
    if (lane == 0) {
#pragma unroll
        for (int j = 0; j < 8; ++j) {
            const float lg = p[j] * ATT_SCALE;
            elog[(size_t)e * 8 + j] = lg;
            atomicMaxF(&Mx[(size_t)d * 8 + j], lg);
        }
    }
}

__global__ __launch_bounds__(256) void k_scatter(const float* __restrict__ elog,
                                                 const float* __restrict__ Mx,
                                                 const float* __restrict__ V,
                                                 const int* __restrict__ src,
                                                 const int* __restrict__ dst,
                                                 float* __restrict__ Sm,
                                                 float* __restrict__ AGG) {
    const int e = blockIdx.x * 8 + (threadIdx.x >> 5);
    const int lane = threadIdx.x & 31;
    const int s = src[e], d = dst[e];
    float ev[8];
#pragma unroll
    for (int j = 0; j < 8; ++j)
        ev[j] = __expf(elog[(size_t)e * 8 + j] - Mx[(size_t)d * 8 + j]);
    if (lane == 0) {
#pragma unroll
        for (int j = 0; j < 8; ++j) atomicAdd(&Sm[(size_t)d * 8 + j], ev[j]);
    }
    const float* v = V + (size_t)s * 256;
    float* ag = AGG + (size_t)d * 256;
#pragma unroll
    for (int j = 0; j < 8; ++j)
        atomicAdd(&ag[j * 32 + lane], ev[j] * v[j * 32 + lane]);
}

// ---------------------------------------------------------------- node combine
__global__ __launch_bounds__(256) void k_combine_sig(const float* __restrict__ AGG,
                                                     const float* __restrict__ Sm,
                                                     const float* __restrict__ S,
                                                     float* __restrict__ out) {
    int i = blockIdx.x * 256 + threadIdx.x;           // over N*256
    int n = i >> 8, hh = (i >> 5) & 7;
    float a = AGG[i] / (Sm[(size_t)n * 8 + hh] + 1e-16f) + S[i];
    out[i] = 1.0f / (1.0f + __expf(-a));
}

__global__ __launch_bounds__(256) void k_final(const float* __restrict__ AGG,
                                               const float* __restrict__ Sm,
                                               const float* __restrict__ S,
                                               const float* __restrict__ Z,
                                               const float* __restrict__ h,
                                               float* __restrict__ out) {
    int i = blockIdx.x * 256 + threadIdx.x;           // over N*256
    int n = i >> 8, hh = (i >> 5) & 7;
    float a = AGG[i] / (Sm[(size_t)n * 8 + hh] + 1e-16f) + S[i];
    float ht = tanhf(a);
    float z = Z[i];
    out[i] = z * h[i] + (1.0f - z) * ht;
}

// ---------------------------------------------------------------- host driver
extern "C" void kernel_launch(void* const* d_in, const int* in_sizes, int n_in,
                              void* d_out, int out_size, void* d_ws, size_t ws_size,
                              hipStream_t stream) {
    (void)in_sizes; (void)n_in; (void)out_size; (void)ws_size;
    const float* x = (const float*)d_in[0];
    const float* h = (const float*)d_in[1];
    const int* ei  = (const int*)d_in[2];
    const int* src = ei;
    const int* dst = ei + NED;

    const float *W[3][4], *B[3][4];
    for (int g = 0; g < 3; ++g)
        for (int p = 0; p < 4; ++p) {
            W[g][p] = (const float*)d_in[3 + g * 8 + p * 2];
            B[g][p] = (const float*)d_in[3 + g * 8 + p * 2 + 1];
        }

    // ---- carve workspace ----
    char* ws = (char*)d_ws;
    size_t off = 0;
    auto carve = [&](size_t bytes) -> void* {
        void* p = ws + off;
        off = (off + bytes + 255) & ~(size_t)255;
        return p;
    };
    unsigned short* hb  = (unsigned short*)carve((size_t)NND * DIN * 2);
    unsigned short* Wb  = (unsigned short*)carve((size_t)DIN * HIDC * 2);
    float* Q   = (float*)carve((size_t)NND * HIDC * 4);
    float* Kb  = (float*)carve((size_t)NND * HIDC * 4);
    float* Vb  = (float*)carve((size_t)NND * HIDC * 4);
    float* Sk  = (float*)carve((size_t)NND * HIDC * 4);
    float* Z   = (float*)carve((size_t)NND * HIDC * 4);
    float* Mx  = (float*)carve((size_t)NND * NH * 4);
    float* Sm  = (float*)carve((size_t)NND * NH * 4);
    float* EL  = (float*)carve((size_t)NED * NH * 4);
    float* AGG = Q;    // alias: Q only needed until k_logits; AGG zeroed after
    float* R   = Kb;   // alias: K only needed until k_logits of r phase

    const int gConcat = (NND * DIN) / 256;        // 40000
    const int gW2bf   = (DIN * HIDC) / 256;       // 512
    const int gGemm   = (NND + 63) / 64;          // 313
    const int gMS     = (NND * NH) / 256;         // 625
    const int gNode   = (NND * HIDC) / 256;       // 20000
    const int gEdge   = NED / 8;                  // 40000

    float* projOut[4] = {Q, Kb, Vb, Sk};

    auto run_tconv = [&](int g) {
        for (int p = 0; p < 4; ++p) {
            k_w2bf<<<gW2bf, 256, 0, stream>>>(W[g][p], Wb);
            k_gemm<<<gGemm, 256, 0, stream>>>(hb, Wb, B[g][p], projOut[p], NND);
        }
        k_init_ms<<<gMS, 256, 0, stream>>>(Mx, Sm);
        k_logits<<<gEdge, 256, 0, stream>>>(Q, Kb, src, dst, EL, Mx);
        k_zero<<<gNode, 256, 0, stream>>>(AGG);   // zero AFTER logits (AGG aliases Q)
        k_scatter<<<gEdge, 256, 0, stream>>>(EL, Mx, Vb, src, dst, Sm, AGG);
    };

    // ---- z gate ----
    k_concat<<<gConcat, 256, 0, stream>>>(x, h, hb);
    run_tconv(0);
    k_combine_sig<<<gNode, 256, 0, stream>>>(AGG, Sm, Sk, Z);

    // ---- r gate ----
    run_tconv(1);
    k_combine_sig<<<gNode, 256, 0, stream>>>(AGG, Sm, Sk, R);

    // ---- candidate ----
    k_hcand<<<gConcat, 256, 0, stream>>>(x, h, R, hb);
    run_tconv(2);
    k_final<<<gNode, 256, 0, stream>>>(AGG, Sm, Sk, Z, h, (float*)d_out);
}
